// GeometryAwareParamHead_83966610637476
// MI455X (gfx1250) — matrix operand order repaired and checked by reading, not compiled, not run on hardware
//
#include <hip/hip_runtime.h>

typedef __attribute__((ext_vector_type(16))) __bf16 v16bf;
typedef __attribute__((ext_vector_type(2)))  __bf16 v2bf;
typedef __attribute__((ext_vector_type(8)))  float  v8f;
typedef __attribute__((ext_vector_type(8)))  int    v8i;

#define B_LATENT 32
#define B_HIDDEN 64
#define B_MAXP   6
#define B_BATCH  524288
#define B_TILES  (B_BATCH / 16)   // 32768
#define TPB_TILES 8               // -> 4096 blocks

// NUM_PARAMS = {4,3,4,3,2,6} packed as nibbles
#define NP_PACK 0x623434

static __device__ __forceinline__ int pack_bf16(float a, float b) {
    v2bf p; p[0] = (__bf16)a; p[1] = (__bf16)b;      // -> v_cvt_pk_bf16_f32
    return __builtin_bit_cast(int, p);
}

// fused pack + ReLU on the packed pair: 2 VALU ops total
static __device__ __forceinline__ int pack_relu_bf16(float a, float b) {
    int r;
    asm("v_cvt_pk_bf16_f32 %0, %1, %2" : "=v"(r) : "v"(a), "v"(b));
    asm("v_pk_max_num_bf16 %0, %1, 0"  : "=v"(r) : "v"(r));
    return r;
}

// sigma: row permutation of W1^T so layer-1 D output lands directly in the
// layer-2 B-fragment layout (no cross-lane movement needed).
static __device__ __forceinline__ int sigma_hcol(int i, int m) {
    return (i & 1) * 8 + (i >> 1) * 32 + m + (m >= 8 ? 8 : 0);
}

__global__ __launch_bounds__(192)
void geom_head_wmma(const float* __restrict__ z,
                    const int*   __restrict__ gt,
                    const float* __restrict__ W1,
                    const float* __restrict__ b1,
                    const float* __restrict__ W2,
                    const float* __restrict__ b2,
                    float* __restrict__ params,
                    float* __restrict__ masks)
{
    const int wave = threadIdx.x >> 5;   // 0..5 == geometry type owned by wave
    const int lane = threadIdx.x & 31;
    const int t    = wave;
    const int n    = lane & 15;          // A-matrix row (hcol/pcol) and B col (batch row)
    const int hi   = lane >> 4;          // half-wave selector
    const int c0   = hi ? 16 : 0;        // A-layout K-chunk base

    // ---------- resident A fragments: sigma-permuted W1^T (16 hcols x 32 latent) ----
    v16bf A1[4];
    {
        const float* w = W1 + t * (B_LATENT * B_HIDDEN);
#pragma unroll
        for (int i = 0; i < 4; ++i) {
            const int hcol = sigma_hcol(i, n);
#pragma unroll
            for (int j = 0; j < 8; ++j) {
                A1[i][j]     = (__bf16)w[(c0 + j)      * B_HIDDEN + hcol];
                A1[i][8 + j] = (__bf16)w[(c0 + 8 + j)  * B_HIDDEN + hcol];
            }
        }
    }
    // ---------- resident A fragments: W2^T padded to 16 rows (16 pcols x 64 hid) ----
    // K index stays the identity thanks to sigma.
    v16bf A2[2];
    {
        const float* w = W2 + t * (B_HIDDEN * B_MAXP);
#pragma unroll
        for (int kf = 0; kf < 2; ++kf) {
#pragma unroll
            for (int j = 0; j < 8; ++j) {
                A2[kf][j]     = (n < B_MAXP) ? (__bf16)w[(kf*32 + c0 + j)      * B_MAXP + n] : (__bf16)0.0f;
                A2[kf][8 + j] = (n < B_MAXP) ? (__bf16)w[(kf*32 + c0 + 8 + j)  * B_MAXP + n] : (__bf16)0.0f;
            }
        }
    }
    // ---------- bias broadcasts (C layout: VGPR r -> M = r + hi*8) ----------
    v8f b1v[4];
#pragma unroll
    for (int i = 0; i < 4; ++i)
#pragma unroll
        for (int r = 0; r < 8; ++r)
            b1v[i][r] = b1[t * B_HIDDEN + sigma_hcol(i, r + hi * 8)];
    v8f b2v;
#pragma unroll
    for (int r = 0; r < 8; ++r) {
        const int p = r + hi * 8;
        b2v[r] = (p < B_MAXP) ? b2[t * B_MAXP + p] : 0.0f;
    }
    const int np_t = (NP_PACK >> (4 * t)) & 0xF;
    const float2 m01 = make_float2(1.0f, 1.0f);                       // np_t >= 2 always
    const float2 m23 = make_float2(np_t > 2 ? 1.f : 0.f, np_t > 3 ? 1.f : 0.f);
    const float2 m45 = make_float2(np_t > 4 ? 1.f : 0.f, np_t > 5 ? 1.f : 0.f);

    for (int it = 0; it < TPB_TILES; ++it) {
        const int tile = blockIdx.x * TPB_TILES + it;
        const int row  = tile * 16 + n;

        // ---- B fragment: z^T (32x16). lane col = batch row; K = hi*16..hi*16+15 ----
        const float4* zp = (const float4*)(z + (size_t)row * B_LATENT + hi * 16);
        const float4 q0 = zp[0], q1 = zp[1], q2 = zp[2], q3 = zp[3];
        v8i zb;
        zb[0] = pack_bf16(q0.x, q0.y); zb[1] = pack_bf16(q0.z, q0.w);
        zb[2] = pack_bf16(q1.x, q1.y); zb[3] = pack_bf16(q1.z, q1.w);
        zb[4] = pack_bf16(q2.x, q2.y); zb[5] = pack_bf16(q2.z, q2.w);
        zb[6] = pack_bf16(q3.x, q3.y); zb[7] = pack_bf16(q3.z, q3.w);
        const v16bf Bz = __builtin_bit_cast(v16bf, zb);

        // ---- layer 1: h^T tiles (sigma-permuted rows); fused relu+pack ----
        int PR[4][4];
#pragma unroll
        for (int i = 0; i < 4; ++i) {
            v8f d = __builtin_amdgcn_wmma_f32_16x16x32_bf16(false, A1[i], false, Bz,
                                                            (short)0, b1v[i], false, false);
#pragma unroll
            for (int j = 0; j < 4; ++j)
                PR[i][j] = pack_relu_bf16(d[2*j], d[2*j + 1]);
        }

        // ---- layer 2: out^T = W2^T x h^T + b2; B frags are direct register
        //      concatenations of the sigma-permuted layer-1 output ----
        v8f acc = b2v;
#pragma unroll
        for (int kf = 0; kf < 2; ++kf) {
            v8i hb;
#pragma unroll
            for (int j = 0; j < 4; ++j) {
                hb[j]     = PR[2*kf][j];
                hb[4 + j] = PR[2*kf + 1][j];
            }
            const v16bf Bh = __builtin_bit_cast(v16bf, hb);
            acc = __builtin_amdgcn_wmma_f32_16x16x32_bf16(false, A2[kf], false, Bh,
                                                          (short)0, acc, false, false);
        }

        // ---- routed store: lane (hi=0) holds all params of its row in acc[0..5];
        //      hi=0 lanes write params, hi=1 lanes write masks for the same row ----
        const int gtv = gt[row];
        if (gtv == t) {
            if (hi == 0) {
                float2* pp = (float2*)(params + (size_t)row * B_MAXP);
                pp[0] = make_float2(acc[0], acc[1]);
                pp[1] = make_float2(acc[2], acc[3]);
                pp[2] = make_float2(acc[4], acc[5]);
            } else {
                float2* mp = (float2*)(masks + (size_t)row * B_MAXP);
                mp[0] = m01;
                mp[1] = m23;
                mp[2] = m45;
            }
        }
    }
}

extern "C" void kernel_launch(void* const* d_in, const int* in_sizes, int n_in,
                              void* d_out, int out_size, void* d_ws, size_t ws_size,
                              hipStream_t stream) {
    const float* z  = (const float*)d_in[0];
    const int*   gt = (const int*)  d_in[1];
    const float* W1 = (const float*)d_in[2];
    const float* b1 = (const float*)d_in[3];
    const float* W2 = (const float*)d_in[4];
    const float* b2 = (const float*)d_in[5];
    float* params = (float*)d_out;
    float* masks  = params + (size_t)B_BATCH * B_MAXP;

    dim3 grid(B_TILES / TPB_TILES);   // 4096
    dim3 block(192);                  // 6 waves, one geometry type each
    geom_head_wmma<<<grid, block, 0, stream>>>(z, gt, W1, b1, W2, b2, params, masks);
}
